// SimpleMemoryNetwork_20243_84765474554646
// MI455X (gfx1250) — matrix operand order repaired and checked by reading, not compiled, not run on hardware
//
#include <hip/hip_runtime.h>
#include <hip/hip_bf16.h>

// ---------------------------------------------------------------------------
// SimpleMemoryNetwork on MI455X (gfx1250, wave32)
//   B=32, T=48, H=512, N_OUT=128, LAM=0.95, ETA=0.5
// Design: bf16 WMMA (v_wmma_f32_16x16x32_bf16) for all dense GEMMs,
// L2-resident 32MB fast-weight tensor A with fused dot+relu+rank1 RMW.
// ---------------------------------------------------------------------------

typedef __bf16 bf16_t;
typedef bf16_t v16bf __attribute__((ext_vector_type(16)));
typedef bf16_t v8bf  __attribute__((ext_vector_type(8)));
typedef float  v8f   __attribute__((ext_vector_type(8)));

#define HDIM   512
#define BATCH  32
#define TSTEPS 48
#define NOUT   128
#define LAMBDA 0.95f
#define ETA    0.5f

// --- WMMA fragment helpers (wave32 layouts per CDNA5 ISA 7.12.2) -----------

// A-matrix 16x32 bf16: lanes 0-15 -> M=lane, K in {k0..k0+7, k0+16..k0+23};
//                      lanes 16-31 -> M=lane-16, K in {k0+8..15, k0+24..31}.
static __device__ __forceinline__ v16bf load_a_frag(const bf16_t* __restrict__ base,
                                                    int m0, int k0) {
  int lane = threadIdx.x & 31;
  int m  = m0 + (lane & 15);
  int kb = k0 + ((lane >> 4) << 3);
  const bf16_t* p = base + m * HDIM + kb;
  v8bf lo = *(const v8bf*)(p);        // K = kb .. kb+7       (16B load)
  v8bf hi = *(const v8bf*)(p + 16);   // K = kb+16 .. kb+23   (16B load)
  v16bf f;
#pragma unroll
  for (int i = 0; i < 8; ++i) { f[i] = lo[i]; f[i + 8] = hi[i]; }
  return f;
}

// B-matrix 32x16 bf16, weight stored transposed [N][K]:
// lanes 0-15 -> N=lane, K in {k0..k0+7, k0+16..k0+23} ; lanes 16-31 -> N=lane-16, K in {k0+8..15, k0+24..31}.
static __device__ __forceinline__ v16bf load_b_frag(const bf16_t* __restrict__ wt,
                                                    int n0, int k0, int ldk) {
  int lane = threadIdx.x & 31;
  int n  = n0 + (lane & 15);
  int kb = k0 + ((lane >> 4) << 3);
  const bf16_t* p = wt + n * ldk + kb;
  v8bf lo = *(const v8bf*)(p);
  v8bf hi = *(const v8bf*)(p + 16);
  v16bf f;
#pragma unroll
  for (int i = 0; i < 8; ++i) { f[i] = lo[i]; f[i + 8] = hi[i]; }
  return f;
}

#define WMMA_BF16(a, b, c) \
  __builtin_amdgcn_wmma_f32_16x16x32_bf16(false, (a), false, (b), (short)0, (c), false, false)

// ---------------------------------------------------------------------------
// Prep kernels
// ---------------------------------------------------------------------------

__global__ void zero_init_kernel(float* __restrict__ A,
                                 bf16_t* __restrict__ r0, bf16_t* __restrict__ a0) {
  size_t tid    = (size_t)blockIdx.x * blockDim.x + threadIdx.x;
  size_t stride = (size_t)gridDim.x * blockDim.x;
  const size_t An = (size_t)BATCH * HDIM * HDIM;
  for (size_t i = tid; i < An; i += stride) A[i] = 0.0f;
  if (tid < (size_t)BATCH * HDIM) { r0[tid] = (bf16_t)0.0f; a0[tid] = (bf16_t)0.0f; }
}

// W[K][N] f32 (row-major)  ->  Wt[N][K] bf16 (row-major)
__global__ void transpose_bf16_kernel(const float* __restrict__ W,
                                      bf16_t* __restrict__ Wt, int K, int N) {
  int idx = blockIdx.x * blockDim.x + threadIdx.x;
  if (idx >= K * N) return;
  int n = idx % N;
  int k = idx / N;
  Wt[n * K + k] = (bf16_t)W[idx];      // coalesced read of W
}

// Pre-pack all T timesteps: x_in (T,B,1025) f32 -> x1_bf, x2_bf (T,B,512), x2last (T,B)
__global__ void pack_x_kernel(const float* __restrict__ x_in,
                              bf16_t* __restrict__ x1, bf16_t* __restrict__ x2,
                              float* __restrict__ x2last) {
  int idx = blockIdx.x * blockDim.x + threadIdx.x;
  if (idx >= TSTEPS * BATCH * HDIM) return;
  int k  = idx & (HDIM - 1);
  int tb = idx >> 9;                                  // t*B + b (contiguous)
  const float* src = x_in + (size_t)tb * (2 * HDIM + 1);
  x1[idx] = (bf16_t)src[k];
  x2[idx] = (bf16_t)src[HDIM + k];
  if (k == 0) x2last[tb] = src[2 * HDIM];
}

// ---------------------------------------------------------------------------
// Per-step kernels
// ---------------------------------------------------------------------------

// blocks 0..7  : r_new = relu(x1 @ W_sr + a @ W_ar)
// blocks 8..15 : a_new = relu(x2[:, :512] @ W_ma[:512] + x2last*W_ma[512] + a @ W_a + r @ W_ra)
// 128 threads = 4 waves; each wave: one 16-col N-tile x both 16-row M-tiles.
__global__ __launch_bounds__(128) void gemm_ra_kernel(
    const bf16_t* __restrict__ x1_bf, const bf16_t* __restrict__ x2_bf,
    const float*  __restrict__ x2last,
    const bf16_t* __restrict__ r_in,  const bf16_t* __restrict__ a_in,
    const bf16_t* __restrict__ Wt_sr, const bf16_t* __restrict__ Wt_ar,
    const bf16_t* __restrict__ Wt_ma, const bf16_t* __restrict__ Wt_a,
    const bf16_t* __restrict__ Wt_ra, const float* __restrict__ Wma_last,
    bf16_t* __restrict__ r_out_bf, bf16_t* __restrict__ a_out_bf,
    float* __restrict__ r_out_f) {
  int wave = threadIdx.x >> 5;
  int lane = threadIdx.x & 31;
  bool do_a = blockIdx.x >= 8;
  int nb = do_a ? (int)blockIdx.x - 8 : (int)blockIdx.x;
  int n0 = nb * 64 + wave * 16;
  v8f c0 = {};
  v8f c1 = {};
  if (!do_a) {
    for (int k0 = 0; k0 < HDIM; k0 += 32) {
      v16bf ax0 = load_a_frag(x1_bf, 0, k0);
      v16bf ax1 = load_a_frag(x1_bf, 16, k0);
      v16bf aa0 = load_a_frag(a_in, 0, k0);
      v16bf aa1 = load_a_frag(a_in, 16, k0);
      v16bf bsr = load_b_frag(Wt_sr, n0, k0, HDIM);
      v16bf bar = load_b_frag(Wt_ar, n0, k0, HDIM);
      c0 = WMMA_BF16(ax0, bsr, c0);
      c0 = WMMA_BF16(aa0, bar, c0);
      c1 = WMMA_BF16(ax1, bsr, c1);
      c1 = WMMA_BF16(aa1, bar, c1);
    }
    int mb = (lane >> 4) << 3;
    int n  = n0 + (lane & 15);
#pragma unroll
    for (int rr = 0; rr < 8; ++rr) {
      int m = mb + rr;
      float v = fmaxf(c0[rr], 0.0f);
      r_out_f [m * HDIM + n] = v;
      r_out_bf[m * HDIM + n] = (bf16_t)v;
      int m2 = 16 + mb + rr;
      float v2 = fmaxf(c1[rr], 0.0f);
      r_out_f [m2 * HDIM + n] = v2;
      r_out_bf[m2 * HDIM + n] = (bf16_t)v2;
    }
  } else {
    for (int k0 = 0; k0 < HDIM; k0 += 32) {
      v16bf ax0 = load_a_frag(x2_bf, 0, k0);
      v16bf ax1 = load_a_frag(x2_bf, 16, k0);
      v16bf aa0 = load_a_frag(a_in, 0, k0);
      v16bf aa1 = load_a_frag(a_in, 16, k0);
      v16bf ar0 = load_a_frag(r_in, 0, k0);
      v16bf ar1 = load_a_frag(r_in, 16, k0);
      v16bf bma = load_b_frag(Wt_ma, n0, k0, HDIM);
      v16bf ba  = load_b_frag(Wt_a,  n0, k0, HDIM);
      v16bf bra = load_b_frag(Wt_ra, n0, k0, HDIM);
      c0 = WMMA_BF16(ax0, bma, c0);
      c0 = WMMA_BF16(aa0, ba,  c0);
      c0 = WMMA_BF16(ar0, bra, c0);
      c1 = WMMA_BF16(ax1, bma, c1);
      c1 = WMMA_BF16(aa1, ba,  c1);
      c1 = WMMA_BF16(ar1, bra, c1);
    }
    int mb = (lane >> 4) << 3;
    int n  = n0 + (lane & 15);
    float wlast = Wma_last[n];        // f32 row 512 of W_ma (ragged K=513 term)
#pragma unroll
    for (int rr = 0; rr < 8; ++rr) {
      int m = mb + rr;
      float v = fmaxf(c0[rr] + x2last[m] * wlast, 0.0f);
      a_out_bf[m * HDIM + n] = (bf16_t)v;
      int m2 = 16 + mb + rr;
      float v2 = fmaxf(c1[rr] + x2last[m2] * wlast, 0.0f);
      a_out_bf[m2 * HDIM + n] = (bf16_t)v2;
    }
  }
}

// o_pre = r_new @ W_ro + a_new @ W_ao   (no activation, f32 out)
__global__ __launch_bounds__(128) void gemm_opre_kernel(
    const bf16_t* __restrict__ r_bf, const bf16_t* __restrict__ a_bf,
    const bf16_t* __restrict__ Wt_ro, const bf16_t* __restrict__ Wt_ao,
    float* __restrict__ o_pre) {
  int wave = threadIdx.x >> 5;
  int lane = threadIdx.x & 31;
  int n0 = blockIdx.x * 64 + wave * 16;
  v8f c0 = {};
  v8f c1 = {};
  for (int k0 = 0; k0 < HDIM; k0 += 32) {
    v16bf ar0 = load_a_frag(r_bf, 0, k0);
    v16bf ar1 = load_a_frag(r_bf, 16, k0);
    v16bf aa0 = load_a_frag(a_bf, 0, k0);
    v16bf aa1 = load_a_frag(a_bf, 16, k0);
    v16bf bro = load_b_frag(Wt_ro, n0, k0, HDIM);
    v16bf bao = load_b_frag(Wt_ao, n0, k0, HDIM);
    c0 = WMMA_BF16(ar0, bro, c0);
    c0 = WMMA_BF16(aa0, bao, c0);
    c1 = WMMA_BF16(ar1, bro, c1);
    c1 = WMMA_BF16(aa1, bao, c1);
  }
  int mb = (lane >> 4) << 3;
  int n  = n0 + (lane & 15);
#pragma unroll
  for (int rr = 0; rr < 8; ++rr) {
    o_pre[(mb + rr) * HDIM + n]      = c0[rr];
    o_pre[(16 + mb + rr) * HDIM + n] = c1[rr];
  }
}

// Fused fast-weight pass: one wave owns one (b,i) row of A (512 f32, L2-hot).
//   dot = A[b,i,:] . r_new[b,:] ;  o = relu(o_pre + dot)
//   A[b,i,:] = LAM*A[b,i,:] + ETA*o*r_new[b,:]
__global__ __launch_bounds__(256) void update_A_kernel(
    float* __restrict__ A, const float* __restrict__ o_pre,
    const float* __restrict__ r_new, bf16_t* __restrict__ o_bf) {
  int wave = threadIdx.x >> 5;
  int lane = threadIdx.x & 31;
  int row = blockIdx.x * 8 + wave;             // 0 .. B*H-1
  int b = row >> 9;
  int i = row & (HDIM - 1);
  float* Arow = A + ((size_t)b << 18) + ((size_t)i << 9);
  const float* rrow = r_new + ((size_t)b << 9);
  __builtin_prefetch(rrow + lane * 4, 0, 3);   // -> global_prefetch_b8

  float4 av[4], rv[4];
  float s = 0.0f;
#pragma unroll
  for (int c = 0; c < 4; ++c) {
    av[c] = ((const float4*)Arow)[c * 32 + lane];   // 512B/instr, coalesced
    rv[c] = ((const float4*)rrow)[c * 32 + lane];
    s += av[c].x * rv[c].x + av[c].y * rv[c].y + av[c].z * rv[c].z + av[c].w * rv[c].w;
  }
  // wave32 butterfly reduction
#pragma unroll
  for (int off = 16; off > 0; off >>= 1) s += __shfl_xor(s, off, 32);

  float o = fmaxf(o_pre[((size_t)b << 9) + i] + s, 0.0f);
  float eo = ETA * o;
#pragma unroll
  for (int c = 0; c < 4; ++c) {
    float4 nv;
    nv.x = LAMBDA * av[c].x + eo * rv[c].x;
    nv.y = LAMBDA * av[c].y + eo * rv[c].y;
    nv.z = LAMBDA * av[c].z + eo * rv[c].z;
    nv.w = LAMBDA * av[c].w + eo * rv[c].w;
    ((float4*)Arow)[c * 32 + lane] = nv;
  }
  if (lane == 0) o_bf[((size_t)b << 9) + i] = (bf16_t)o;
}

// y = o @ W_y   (32 x 128, K=512)
__global__ __launch_bounds__(128) void gemm_y_kernel(
    const bf16_t* __restrict__ o_bf, const bf16_t* __restrict__ Wt_y,
    float* __restrict__ y) {
  int wave = threadIdx.x >> 5;
  int lane = threadIdx.x & 31;
  int n0 = blockIdx.x * 64 + wave * 16;
  v8f c0 = {};
  v8f c1 = {};
  for (int k0 = 0; k0 < HDIM; k0 += 32) {
    v16bf a0 = load_a_frag(o_bf, 0, k0);
    v16bf a1 = load_a_frag(o_bf, 16, k0);
    v16bf b  = load_b_frag(Wt_y, n0, k0, HDIM);
    c0 = WMMA_BF16(a0, b, c0);
    c1 = WMMA_BF16(a1, b, c1);
  }
  int mb = (lane >> 4) << 3;
  int n  = n0 + (lane & 15);
#pragma unroll
  for (int rr = 0; rr < 8; ++rr) {
    y[(mb + rr) * NOUT + n]      = c0[rr];
    y[(16 + mb + rr) * NOUT + n] = c1[rr];
  }
}

// ---------------------------------------------------------------------------
// Host launch (graph-capture safe: only stream kernel launches)
// ---------------------------------------------------------------------------

extern "C" void kernel_launch(void* const* d_in, const int* in_sizes, int n_in,
                              void* d_out, int out_size, void* d_ws, size_t ws_size,
                              hipStream_t stream) {
  (void)in_sizes; (void)n_in; (void)out_size; (void)ws_size;
  const float* x_in = (const float*)d_in[0];
  const float* W_sr = (const float*)d_in[1];
  const float* W_ma = (const float*)d_in[2];   // (513, 512)
  const float* W_ro = (const float*)d_in[3];
  const float* W_ao = (const float*)d_in[4];
  const float* W_ar = (const float*)d_in[5];
  const float* W_ra = (const float*)d_in[6];
  const float* W_a  = (const float*)d_in[7];
  const float* W_y  = (const float*)d_in[8];   // (512, 128)

  char* base = (char*)d_ws;
  size_t off = 0;
  auto wsalloc = [&](size_t bytes) -> void* {
    void* p = base + off;
    off = (off + bytes + 255) & ~(size_t)255;
    return p;
  };

  float*  A      = (float*) wsalloc((size_t)BATCH * HDIM * HDIM * 4);   // 32 MB, L2-resident
  bf16_t* Wt_sr  = (bf16_t*)wsalloc((size_t)HDIM * HDIM * 2);
  bf16_t* Wt_ar  = (bf16_t*)wsalloc((size_t)HDIM * HDIM * 2);
  bf16_t* Wt_ma  = (bf16_t*)wsalloc((size_t)HDIM * HDIM * 2);           // first 512 K-rows
  bf16_t* Wt_a   = (bf16_t*)wsalloc((size_t)HDIM * HDIM * 2);
  bf16_t* Wt_ra  = (bf16_t*)wsalloc((size_t)HDIM * HDIM * 2);
  bf16_t* Wt_ro  = (bf16_t*)wsalloc((size_t)HDIM * HDIM * 2);
  bf16_t* Wt_ao  = (bf16_t*)wsalloc((size_t)HDIM * HDIM * 2);
  bf16_t* Wt_y   = (bf16_t*)wsalloc((size_t)NOUT * HDIM * 2);
  bf16_t* x1_bf  = (bf16_t*)wsalloc((size_t)TSTEPS * BATCH * HDIM * 2);
  bf16_t* x2_bf  = (bf16_t*)wsalloc((size_t)TSTEPS * BATCH * HDIM * 2);
  float*  x2last = (float*) wsalloc((size_t)TSTEPS * BATCH * 4);
  bf16_t* r_st0  = (bf16_t*)wsalloc((size_t)BATCH * HDIM * 2);
  bf16_t* r_st1  = (bf16_t*)wsalloc((size_t)BATCH * HDIM * 2);
  bf16_t* a_st0  = (bf16_t*)wsalloc((size_t)BATCH * HDIM * 2);
  bf16_t* a_st1  = (bf16_t*)wsalloc((size_t)BATCH * HDIM * 2);
  float*  r_f32  = (float*) wsalloc((size_t)BATCH * HDIM * 4);
  float*  o_pre  = (float*) wsalloc((size_t)BATCH * HDIM * 4);
  bf16_t* o_bf   = (bf16_t*)wsalloc((size_t)BATCH * HDIM * 2);

  bf16_t* r_st[2] = {r_st0, r_st1};
  bf16_t* a_st[2] = {a_st0, a_st1};

  // --- prep: zero state + A, transpose weights to bf16 [N][K], pack x ------
  zero_init_kernel<<<2048, 256, 0, stream>>>(A, r_st[0], a_st[0]);

  int nW  = HDIM * HDIM;
  int gW  = (nW + 255) / 256;
  transpose_bf16_kernel<<<gW, 256, 0, stream>>>(W_sr, Wt_sr, HDIM, HDIM);
  transpose_bf16_kernel<<<gW, 256, 0, stream>>>(W_ar, Wt_ar, HDIM, HDIM);
  transpose_bf16_kernel<<<gW, 256, 0, stream>>>(W_ma, Wt_ma, HDIM, HDIM); // rows 0..511 only
  transpose_bf16_kernel<<<gW, 256, 0, stream>>>(W_a,  Wt_a,  HDIM, HDIM);
  transpose_bf16_kernel<<<gW, 256, 0, stream>>>(W_ra, Wt_ra, HDIM, HDIM);
  transpose_bf16_kernel<<<gW, 256, 0, stream>>>(W_ro, Wt_ro, HDIM, HDIM);
  transpose_bf16_kernel<<<gW, 256, 0, stream>>>(W_ao, Wt_ao, HDIM, HDIM);
  int nY = HDIM * NOUT;
  transpose_bf16_kernel<<<(nY + 255) / 256, 256, 0, stream>>>(W_y, Wt_y, HDIM, NOUT);

  int nX = TSTEPS * BATCH * HDIM;
  pack_x_kernel<<<(nX + 255) / 256, 256, 0, stream>>>(x_in, x1_bf, x2_bf, x2last);

  // --- sequential scan over T (stream-ordered) ------------------------------
  const float* Wma_last = W_ma + (size_t)HDIM * HDIM;  // row 512 of W_ma, f32
  for (int t = 0; t < TSTEPS; ++t) {
    const bf16_t* x1_t  = x1_bf + (size_t)t * BATCH * HDIM;
    const bf16_t* x2_t  = x2_bf + (size_t)t * BATCH * HDIM;
    const float*  x2l_t = x2last + (size_t)t * BATCH;
    bf16_t* r_in  = r_st[t & 1];
    bf16_t* a_in  = a_st[t & 1];
    bf16_t* r_out = r_st[(t + 1) & 1];
    bf16_t* a_out = a_st[(t + 1) & 1];

    gemm_ra_kernel<<<16, 128, 0, stream>>>(
        x1_t, x2_t, x2l_t, r_in, a_in,
        Wt_sr, Wt_ar, Wt_ma, Wt_a, Wt_ra, Wma_last,
        r_out, a_out, r_f32);

    gemm_opre_kernel<<<8, 128, 0, stream>>>(r_out, a_out, Wt_ro, Wt_ao, o_pre);

    update_A_kernel<<<2048, 256, 0, stream>>>(A, o_pre, r_f32, o_bf);
  }

  // --- y = o_last @ W_y -----------------------------------------------------
  gemm_y_kernel<<<2, 128, 0, stream>>>(o_bf, Wt_y, (float*)d_out);
}
